// Decoder_54460185313885
// MI455X (gfx1250) — compile-verified, matrix operand order checked by reading, not run on hardware
//
#include <hip/hip_runtime.h>
#include <cstdint>
#include <cstddef>

// ---------------- problem constants ----------------
#define B_      64
#define TENC_   800
#define EPROJS_ 512
#define D_      512
#define ODIM_   5000
#define L_      128
#define ATT_    320
#define SOS_    (ODIM_ - 1)
#define EOS_    (ODIM_ - 1)
#define LP1_    (L_ + 1)          // 129
#define NROWS_  (B_ * LP1_)       // 8256 = 516 * 16
#define NPADV_  5008              // ODIM padded to 16
#define NTILES_ (NPADV_ / 16)     // 313

// ---------------- WMMA types (gfx1250, wave32) ----------------
typedef __attribute__((ext_vector_type(16))) __bf16        v16bf;
typedef __attribute__((ext_vector_type(8)))  float         v8f;
typedef __attribute__((ext_vector_type(8)))  unsigned short u16x8;

union bfrag { v16bf f; u16x8 h[2]; };

__device__ __forceinline__ v8f wmma_bf16(const bfrag& a, const bfrag& b, v8f c) {
  // v_wmma_f32_16x16x32_bf16  (neg_a, A, neg_b, B, c_mod, C, reuse_a, reuse_b)
  return __builtin_amdgcn_wmma_f32_16x16x32_bf16(false, a.f, false, b.f, (short)0, c, false, false);
}

// load a 16x32 bf16 fragment (A- or BT-layout) from a row-major [rows][K] matrix:
// lane holds row (lane&15); elements = two contiguous 16B runs at k+8g and k+16+8g
__device__ __forceinline__ void load_frag(bfrag& d, const unsigned short* rowbase_plus8g, int kbase) {
  const unsigned short* p = rowbase_plus8g + kbase;
  d.h[0] = *(const u16x8*)(p);
  d.h[1] = *(const u16x8*)(p + 16);
}

// ---------------- bf16 scalar helpers ----------------
__device__ __forceinline__ unsigned short f2bf(float x) {
  union { float f; unsigned int u; } v; v.f = x;
  unsigned int r = v.u + 0x7FFFu + ((v.u >> 16) & 1u);
  return (unsigned short)(r >> 16);
}
__device__ __forceinline__ float bf2f(unsigned short h) {
  union { unsigned int u; float f; } v; v.u = ((unsigned int)h) << 16;
  return v.f;
}
__device__ __forceinline__ float sigmoidf_(float x) { return 1.0f / (1.0f + expf(-x)); }

// =====================================================================
// elementwise prep kernels
// =====================================================================
__global__ void zero_f32_kernel(float* p, int n) {
  for (int i = blockIdx.x * blockDim.x + threadIdx.x; i < n; i += gridDim.x * blockDim.x) p[i] = 0.0f;
}
__global__ void zero_u16_kernel(unsigned short* p, int n) {
  for (int i = blockIdx.x * blockDim.x + threadIdx.x; i < n; i += gridDim.x * blockDim.x) p[i] = 0;
}

__global__ void prep_hm_kernel(const float* __restrict__ hpad, const int* __restrict__ hlen,
                               unsigned short* __restrict__ hm) {
  const int total = B_ * TENC_ * EPROJS_;
  for (int i = blockIdx.x * blockDim.x + threadIdx.x; i < total; i += gridDim.x * blockDim.x) {
    int bt = i >> 9;
    int t  = bt % TENC_;
    int b  = bt / TENC_;
    hm[i] = (t < hlen[b]) ? f2bf(hpad[i]) : (unsigned short)0;
  }
}

__global__ void prep_wcat0_kernel(const float* __restrict__ Wih, const float* __restrict__ Whh,
                                  unsigned short* __restrict__ out) {
  const int total = 2048 * 1536;
  for (int i = blockIdx.x * blockDim.x + threadIdx.x; i < total; i += gridDim.x * blockDim.x) {
    int col = i % 1536, row = i / 1536;
    float v = (col < 1024) ? Wih[row * 1024 + col] : Whh[row * 512 + (col - 1024)];
    out[i] = f2bf(v);
  }
}
__global__ void prep_wcat1_kernel(const float* __restrict__ Wih, const float* __restrict__ Whh,
                                  unsigned short* __restrict__ out) {
  const int total = 2048 * 1024;
  for (int i = blockIdx.x * blockDim.x + threadIdx.x; i < total; i += gridDim.x * blockDim.x) {
    int col = i & 1023, row = i >> 10;
    float v = (col < 512) ? Wih[row * 512 + col] : Whh[row * 512 + (col - 512)];
    out[i] = f2bf(v);
  }
}
__global__ void prep_waencT_kernel(const float* __restrict__ Wa, unsigned short* __restrict__ out) {
  const int total = ATT_ * 512;
  for (int i = blockIdx.x * blockDim.x + threadIdx.x; i < total; i += gridDim.x * blockDim.x) {
    int k = i & 511, a = i >> 9;
    out[i] = f2bf(Wa[k * ATT_ + a]);
  }
}
__global__ void prep_wout_kernel(const float* __restrict__ W, unsigned short* __restrict__ out) {
  const int total = NPADV_ * 512;
  for (int i = blockIdx.x * blockDim.x + threadIdx.x; i < total; i += gridDim.x * blockDim.x) {
    int row = i >> 9;
    out[i] = (row < ODIM_) ? f2bf(W[i]) : (unsigned short)0;
  }
}
__global__ void prep_bias_kernel(const float* __restrict__ bi, const float* __restrict__ bh,
                                 float* __restrict__ out) {
  for (int i = blockIdx.x * blockDim.x + threadIdx.x; i < 2048; i += gridDim.x * blockDim.x)
    out[i] = bi[i] + bh[i];
}
__global__ void gather_embed_kernel(const int* __restrict__ ys, const float* __restrict__ embed,
                                    unsigned short* __restrict__ eys) {
  const int total = LP1_ * B_ * D_;
  for (int i = blockIdx.x * blockDim.x + threadIdx.x; i < total; i += gridDim.x * blockDim.x) {
    int u = i & 511;
    int bl = i >> 9;
    int b = bl & 63;
    int l = bl >> 6;
    int tok = (l == 0) ? SOS_ : ys[b * L_ + (l - 1)];
    eys[i] = f2bf(embed[(size_t)tok * D_ + u]);
  }
}

// =====================================================================
// enc_pre = tanh(hm @ Wa_enc)   M=51200, K=512, N=320 -> bf16
// one wave per 16-row M tile; A cached in regs; B double-buffered
// =====================================================================
__global__ void __launch_bounds__(256) encpre_gemm_kernel(const unsigned short* __restrict__ hm,
                                                          const unsigned short* __restrict__ WaT,
                                                          unsigned short* __restrict__ out) {
  const int tid  = threadIdx.x;
  const int wave = tid >> 5;
  const int lane = tid & 31;
  const int l15  = lane & 15;
  const int g    = lane >> 4;
  const int mtile = blockIdx.x * 8 + wave;      // 0..3199

  const unsigned short* abase = hm + ((size_t)(mtile * 16 + l15)) * 512 + 8 * g;
  bfrag a[16];
  #pragma unroll
  for (int kc = 0; kc < 16; ++kc) load_frag(a[kc], abase, kc * 32);

  for (int nt = 0; nt < ATT_ / 16; ++nt) {
    v8f acc = {0.f, 0.f, 0.f, 0.f, 0.f, 0.f, 0.f, 0.f};
    const unsigned short* brow = WaT + (size_t)(nt * 16 + l15) * 512 + 8 * g;
    bfrag b0, b1;
    load_frag(b0, brow, 0);
    #pragma unroll
    for (int kc = 0; kc < 16; kc += 2) {
      load_frag(b1, brow, (kc + 1) * 32);
      acc = wmma_bf16(a[kc], b0, acc);
      if (kc + 2 < 16) load_frag(b0, brow, (kc + 2) * 32);
      acc = wmma_bf16(a[kc + 1], b1, acc);
    }
    const int ncol = nt * 16 + l15;
    #pragma unroll
    for (int r = 0; r < 8; ++r) {
      int mrow = mtile * 16 + r + 8 * g;
      out[(size_t)mrow * ATT_ + ncol] = f2bf(tanhf(acc[r]));
    }
  }
}

// =====================================================================
// attention: d=tanh(z0@Wa_dec); e=<d,enc_pre>; softmax(mask); att_c=w@hm
// one block per batch row; 16B-vectorized bf16 loads
// =====================================================================
__global__ void __launch_bounds__(256) att_kernel(const float* __restrict__ z0f,
                                                  const float* __restrict__ Wa_dec,
                                                  const unsigned short* __restrict__ encpre,
                                                  const unsigned short* __restrict__ hm,
                                                  const int* __restrict__ hlen,
                                                  unsigned short* __restrict__ attc) {
  __shared__ float z0sh[D_];
  __shared__ float dsh[ATT_];
  __shared__ float wsh[TENC_];
  __shared__ float red[256];
  __shared__ float cred[4][EPROJS_];

  const int b = blockIdx.x;
  const int tid = threadIdx.x;
  const int len = hlen[b];

  for (int k = tid; k < D_; k += 256) z0sh[k] = z0f[b * D_ + k];
  __syncthreads();

  for (int a = tid; a < ATT_; a += 256) {
    float s = 0.f;
    for (int k = 0; k < D_; ++k) s += z0sh[k] * Wa_dec[k * ATT_ + a];
    dsh[a] = tanhf(s);
  }
  __syncthreads();

  // energies, vectorized 8-wide bf16 loads
  float lmax = -1e30f;
  for (int t = tid; t < TENC_; t += 256) {
    float e;
    if (t < len) {
      const u16x8* ep = (const u16x8*)(encpre + ((size_t)b * TENC_ + t) * ATT_);
      e = 0.f;
      #pragma unroll 5
      for (int a8 = 0; a8 < ATT_ / 8; ++a8) {
        u16x8 v = ep[a8];
        #pragma unroll
        for (int j = 0; j < 8; ++j) e += dsh[a8 * 8 + j] * bf2f(v[j]);
      }
    } else {
      e = -1e30f;
    }
    wsh[t] = e;
    lmax = fmaxf(lmax, e);
  }
  red[tid] = lmax; __syncthreads();
  for (int s = 128; s > 0; s >>= 1) { if (tid < s) red[tid] = fmaxf(red[tid], red[tid + s]); __syncthreads(); }
  const float gmax = red[0];
  __syncthreads();

  float lsum = 0.f;
  for (int t = tid; t < TENC_; t += 256) {
    float wv = (t < len) ? expf(wsh[t] - gmax) : 0.f;
    wsh[t] = wv;
    lsum += wv;
  }
  red[tid] = lsum; __syncthreads();
  for (int s = 128; s > 0; s >>= 1) { if (tid < s) red[tid] += red[tid + s]; __syncthreads(); }
  const float inv = 1.0f / red[0];
  __syncthreads();

  // att_c: 4 t-groups x 64 e-blocks, 16B coalesced loads, LDS reduce
  {
    const int eb = tid & 63;        // e block: covers e = eb*8 .. eb*8+7
    const int tg = tid >> 6;        // 0..3
    float acc8[8];
    #pragma unroll
    for (int j = 0; j < 8; ++j) acc8[j] = 0.f;
    const unsigned short* hbase = hm + (size_t)b * TENC_ * EPROJS_ + eb * 8;
    for (int t = tg; t < len; t += 4) {
      const float wv = wsh[t];
      u16x8 hv = *(const u16x8*)(hbase + (size_t)t * EPROJS_);
      #pragma unroll
      for (int j = 0; j < 8; ++j) acc8[j] += wv * bf2f(hv[j]);
    }
    #pragma unroll
    for (int j = 0; j < 8; ++j) cred[tg][eb * 8 + j] = acc8[j];
    __syncthreads();
    for (int e = tid; e < EPROJS_; e += 256)
      attc[b * EPROJS_ + e] = f2bf((cred[0][e] + cred[1][e] + cred[2][e] + cred[3][e]) * inv);
  }
}

// =====================================================================
// fused LSTM cell: g = Xcat @ Wcat^T + bias ; gates ; c/h update
// grid 4 blocks x 512 thr (16 waves); wave owns all 4 gates of its 32
// hidden units -> gates fused in registers. A+B double-buffered.
// =====================================================================
__global__ void __launch_bounds__(512) lstm_gemm_kernel(const unsigned short* __restrict__ segA0,
                                                        const unsigned short* __restrict__ segA1,
                                                        const unsigned short* __restrict__ segA2,
                                                        int Kdim,
                                                        const unsigned short* __restrict__ W,
                                                        const float* __restrict__ bias,
                                                        float* __restrict__ c_state,
                                                        float* __restrict__ h_f32,        // nullable
                                                        unsigned short* __restrict__ h_bf,
                                                        unsigned short* __restrict__ zall, // nullable
                                                        int step_l) {
  const int tid  = threadIdx.x;
  const int wave = tid >> 5;     // 0..15
  const int lane = tid & 31;
  const int l15  = lane & 15;
  const int g    = lane >> 4;
  const int mtile = blockIdx.x;  // 0..3
  const int arow  = mtile * 16 + l15;

  const unsigned short* segs[3] = {segA0, segA1, segA2};

  v8f acc[8];
  #pragma unroll
  for (int t = 0; t < 8; ++t) acc[t] = (v8f){0.f, 0.f, 0.f, 0.f, 0.f, 0.f, 0.f, 0.f};

  const unsigned short* wrow[8];
  const int ug0 = wave, ug1 = wave + 16;
  #pragma unroll
  for (int t = 0; t < 8; ++t) {
    int ug = (t < 4) ? ug0 : ug1;
    int gate = t & 3;
    int nt = ug + 32 * gate;                      // output cols [nt*16, nt*16+16)
    wrow[t] = W + (size_t)(nt * 16 + l15) * Kdim + 8 * g;
  }

  const int kchunks = Kdim >> 5;

  bfrag a0, a1, b0[8], b1[8];
  #define LOAD_A(dst, kc) do {                                                   \
      const int kb_ = (kc) << 5;                                                 \
      load_frag(dst, segs[kb_ >> 9] + (size_t)arow * 512 + 8 * g, kb_ & 511);    \
    } while (0)
  #define LOAD_B(dst, kc) do {                                                   \
      const int kb_ = (kc) << 5;                                                 \
      _Pragma("unroll")                                                          \
      for (int t_ = 0; t_ < 8; ++t_) load_frag(dst[t_], wrow[t_], kb_);          \
    } while (0)

  LOAD_A(a0, 0); LOAD_B(b0, 0);
  for (int kc = 0; kc + 2 <= kchunks; kc += 2) {
    LOAD_A(a1, kc + 1); LOAD_B(b1, kc + 1);
    #pragma unroll
    for (int t = 0; t < 8; ++t) acc[t] = wmma_bf16(a0, b0[t], acc[t]);
    if (kc + 2 < kchunks) { LOAD_A(a0, kc + 2); LOAD_B(b0, kc + 2); }
    #pragma unroll
    for (int t = 0; t < 8; ++t) acc[t] = wmma_bf16(a1, b1[t], acc[t]);
  }
  #undef LOAD_A
  #undef LOAD_B

  // gate fusion: lane's column u, rows r + 8g
  #pragma unroll
  for (int half = 0; half < 2; ++half) {
    const int ug = half ? ug1 : ug0;
    const int u  = ug * 16 + l15;                 // hidden unit index 0..511
    const v8f ai = acc[half * 4 + 0];
    const v8f af = acc[half * 4 + 1];
    const v8f ag = acc[half * 4 + 2];
    const v8f ao = acc[half * 4 + 3];
    const float bi = bias[u], bf_ = bias[512 + u], bg = bias[1024 + u], bo = bias[1536 + u];
    #pragma unroll
    for (int r = 0; r < 8; ++r) {
      const int row = mtile * 16 + r + 8 * g;     // batch index
      const float iv = sigmoidf_(ai[r] + bi);
      const float fv = sigmoidf_(af[r] + bf_);
      const float gv = tanhf(ag[r] + bg);
      const float ov = sigmoidf_(ao[r] + bo);
      const float c2 = fv * c_state[row * D_ + u] + iv * gv;
      const float h2 = ov * tanhf(c2);
      c_state[row * D_ + u] = c2;
      h_bf[row * D_ + u] = f2bf(h2);
      if (h_f32) h_f32[row * D_ + u] = h2;
      if (zall)  zall[((size_t)row * LP1_ + step_l) * D_ + u] = f2bf(h2);
    }
  }
}

// =====================================================================
// logits + log_softmax + nll + argmax-acc, fused with online logsumexp.
// grid 516 blocks (16 rows each) x 256 thr (8 waves). A-tile staged into
// LDS with gfx1250 async-to-LDS DMA; B double-buffered from L2.
// =====================================================================
__global__ void __launch_bounds__(256) logits_loss_kernel(const unsigned short* __restrict__ zall,
                                                          const unsigned short* __restrict__ Wout,
                                                          const float* __restrict__ b_out,
                                                          const int* __restrict__ ys,
                                                          float* __restrict__ accum) {
  __shared__ __align__(16) unsigned short Ash[16 * 512];
  __shared__ int   tgt_id[16];
  __shared__ float tgt_logit[16];
  __shared__ float lm[16][128];
  __shared__ float ls[16][128];
  __shared__ float lav[16][128];
  __shared__ int   lai[16][128];

  const int tid  = threadIdx.x;
  const int wave = tid >> 5;
  const int lane = tid & 31;
  const int l15  = lane & 15;
  const int g    = lane >> 4;
  const size_t rowbase = (size_t)blockIdx.x * 16;

  // ---- async DMA the 16x512 bf16 A-tile into LDS (ASYNCcnt-tracked) ----
  {
    const unsigned short* gsrc = zall + rowbase * 512;
    #pragma unroll
    for (int it = 0; it < 4; ++it) {                 // 1024 x 16B chunks / 256 thr
      const int i = tid + it * 256;
      unsigned lds_addr = (unsigned)(uintptr_t)(&Ash[i * 8]);
      unsigned long long gaddr = (unsigned long long)(uintptr_t)(gsrc + i * 8);
      asm volatile("global_load_async_to_lds_b128 %0, %1, off"
                   :: "v"(lds_addr), "v"(gaddr) : "memory");
    }
    asm volatile("s_wait_asynccnt 0x0" ::: "memory");
  }
  if (tid < 16) {
    const int row = (int)rowbase + tid;
    const int b = row / LP1_, l = row % LP1_;
    tgt_id[tid] = (l < L_) ? ys[b * L_ + l] : EOS_;
    tgt_logit[tid] = 0.f;
  }
  __syncthreads();

  float m[8], s[8], av[8];
  int   ai_[8];
  #pragma unroll
  for (int r = 0; r < 8; ++r) { m[r] = -1e38f; s[r] = 0.f; av[r] = -1e38f; ai_[r] = 0; }

  const unsigned short* ashbase = &Ash[l15 * 512 + 8 * g];

  for (int nt = wave; nt < NTILES_; nt += 8) {
    v8f acc = {0.f, 0.f, 0.f, 0.f, 0.f, 0.f, 0.f, 0.f};
    const unsigned short* brow = Wout + (size_t)(nt * 16 + l15) * 512 + 8 * g;
    bfrag bb0, bb1, af;
    load_frag(bb0, brow, 0);
    #pragma unroll
    for (int kc = 0; kc < 16; kc += 2) {
      load_frag(bb1, brow, (kc + 1) * 32);
      load_frag(af, ashbase, kc * 32);
      acc = wmma_bf16(af, bb0, acc);
      if (kc + 2 < 16) load_frag(bb0, brow, (kc + 2) * 32);
      load_frag(af, ashbase, (kc + 1) * 32);
      acc = wmma_bf16(af, bb1, acc);
    }
    const int n = nt * 16 + l15;
    const bool valid = (n < ODIM_);
    const float bo = valid ? b_out[n] : 0.f;
    #pragma unroll
    for (int r = 0; r < 8; ++r) {
      if (valid) {
        const float lg = acc[r] + bo;
        if (lg > m[r]) { s[r] = s[r] * expf(m[r] - lg) + 1.f; m[r] = lg; }
        else           { s[r] += expf(lg - m[r]); }
        if (lg > av[r]) { av[r] = lg; ai_[r] = n; }
        if (n == tgt_id[r + 8 * g]) tgt_logit[r + 8 * g] = lg;
      }
    }
  }

  const int col = wave * 16 + l15;
  #pragma unroll
  for (int r = 0; r < 8; ++r) {
    const int rl = r + 8 * g;
    lm[rl][col] = m[r]; ls[rl][col] = s[r]; lav[rl][col] = av[r]; lai[rl][col] = ai_[r];
  }
  __syncthreads();

  if (tid < 16) {
    float M = -1e38f, S = 0.f, AV = -1e38f;
    int AI = 0;
    for (int c = 0; c < 128; ++c) {
      const float mm = lm[tid][c];
      if (mm > -1e37f) {
        if (mm > M) { S = S * expf(M - mm) + ls[tid][c]; M = mm; }
        else        { S += ls[tid][c] * expf(mm - M); }
      }
      const float v = lav[tid][c];
      if (v > AV || (v == AV && lai[tid][c] < AI)) { AV = v; AI = lai[tid][c]; }
    }
    const float lse = M + logf(S);
    const float nll = lse - tgt_logit[tid];
    atomicAdd(&accum[0], nll);
    atomicAdd(&accum[1], (AI == tgt_id[tid]) ? 1.0f : 0.0f);
  }
}

__global__ void finalize_kernel(const float* __restrict__ accum, float* __restrict__ out) {
  if (threadIdx.x == 0 && blockIdx.x == 0) {
    out[0] = accum[0] / (float)NROWS_ * (float)L_;   // mean * L
    out[1] = accum[1] / (float)NROWS_;
  }
}

// =====================================================================
// host launcher
// =====================================================================
static inline uint8_t* align256(uint8_t* p) {
  return (uint8_t*)(((uintptr_t)p + 255) & ~(uintptr_t)255);
}
static inline unsigned gb(size_t n) { size_t b = (n + 255) / 256; return (unsigned)(b > 131072 ? 131072 : b); }

extern "C" void kernel_launch(void* const* d_in, const int* in_sizes, int n_in,
                              void* d_out, int out_size, void* d_ws, size_t ws_size,
                              hipStream_t stream) {
  const float* hpad   = (const float*)d_in[0];
  const int*   hlen   = (const int*)d_in[1];
  const int*   ys     = (const int*)d_in[2];
  const float* embed  = (const float*)d_in[3];
  const float* W_ih0  = (const float*)d_in[4];
  const float* W_hh0  = (const float*)d_in[5];
  const float* b_ih0  = (const float*)d_in[6];
  const float* b_hh0  = (const float*)d_in[7];
  const float* W_ih1  = (const float*)d_in[8];
  const float* W_hh1  = (const float*)d_in[9];
  const float* b_ih1  = (const float*)d_in[10];
  const float* b_hh1  = (const float*)d_in[11];
  const float* Wa_enc = (const float*)d_in[12];
  const float* Wa_dec = (const float*)d_in[13];
  const float* W_out  = (const float*)d_in[14];
  const float* b_out  = (const float*)d_in[15];
  float* out = (float*)d_out;

  // ---- carve workspace ----
  uint8_t* p = (uint8_t*)d_ws;
  #define CARVE(name, type, count) type* name = (type*)align256(p); p = (uint8_t*)(name + (size_t)(count))
  CARVE(hm_bf,     unsigned short, (size_t)B_ * TENC_ * EPROJS_);
  CARVE(encpre_bf, unsigned short, (size_t)B_ * TENC_ * ATT_);
  CARVE(waencT,    unsigned short, (size_t)ATT_ * 512);
  CARVE(wcat0,     unsigned short, (size_t)2048 * 1536);
  CARVE(wcat1,     unsigned short, (size_t)2048 * 1024);
  CARVE(wout_bf,   unsigned short, (size_t)NPADV_ * 512);
  CARVE(eys_bf,    unsigned short, (size_t)LP1_ * B_ * D_);
  CARVE(zall_bf,   unsigned short, (size_t)B_ * LP1_ * D_);
  CARVE(attc_bf,   unsigned short, (size_t)B_ * EPROJS_);
  CARVE(z0_bf,     unsigned short, (size_t)B_ * D_);
  CARVE(z1_bf,     unsigned short, (size_t)B_ * D_);
  CARVE(bias0,     float,          2048);
  CARVE(bias1,     float,          2048);
  CARVE(z0f,       float,          (size_t)B_ * D_);
  CARVE(c0,        float,          (size_t)B_ * D_);
  CARVE(c1,        float,          (size_t)B_ * D_);
  CARVE(accum,     float,          8);
  #undef CARVE

  // ---- init state (every call: deterministic) ----
  zero_f32_kernel<<<gb(B_ * D_), 256, 0, stream>>>(z0f, B_ * D_);
  zero_f32_kernel<<<gb(B_ * D_), 256, 0, stream>>>(c0, B_ * D_);
  zero_f32_kernel<<<gb(B_ * D_), 256, 0, stream>>>(c1, B_ * D_);
  zero_u16_kernel<<<gb(B_ * D_), 256, 0, stream>>>(z0_bf, B_ * D_);
  zero_u16_kernel<<<gb(B_ * D_), 256, 0, stream>>>(z1_bf, B_ * D_);
  zero_f32_kernel<<<1, 32, 0, stream>>>(accum, 8);

  // ---- weight / input prep ----
  prep_hm_kernel<<<gb((size_t)B_ * TENC_ * EPROJS_), 256, 0, stream>>>(hpad, hlen, hm_bf);
  prep_wcat0_kernel<<<gb(2048 * 1536), 256, 0, stream>>>(W_ih0, W_hh0, wcat0);
  prep_wcat1_kernel<<<gb(2048 * 1024), 256, 0, stream>>>(W_ih1, W_hh1, wcat1);
  prep_waencT_kernel<<<gb(ATT_ * 512), 256, 0, stream>>>(Wa_enc, waencT);
  prep_wout_kernel<<<gb((size_t)NPADV_ * 512), 256, 0, stream>>>(W_out, wout_bf);
  prep_bias_kernel<<<8, 256, 0, stream>>>(b_ih0, b_hh0, bias0);
  prep_bias_kernel<<<8, 256, 0, stream>>>(b_ih1, b_hh1, bias1);
  gather_embed_kernel<<<gb((size_t)LP1_ * B_ * D_), 256, 0, stream>>>(ys, embed, eys_bf);

  // ---- enc_pre GEMM (WMMA) ----
  encpre_gemm_kernel<<<(B_ * TENC_ / 16) / 8, 256, 0, stream>>>(hm_bf, waencT, encpre_bf);

  // ---- sequential decoder: 129 steps x (attention, LSTM0, LSTM1) ----
  for (int l = 0; l < LP1_; ++l) {
    att_kernel<<<B_, 256, 0, stream>>>(z0f, Wa_dec, encpre_bf, hm_bf, hlen, attc_bf);
    lstm_gemm_kernel<<<4, 512, 0, stream>>>(eys_bf + (size_t)l * B_ * D_, attc_bf, z0_bf,
                                            1536, wcat0, bias0, c0,
                                            z0f, z0_bf, (unsigned short*)nullptr, 0);
    lstm_gemm_kernel<<<4, 512, 0, stream>>>(z0_bf, z1_bf, (const unsigned short*)nullptr,
                                            1024, wcat1, bias1, c1,
                                            (float*)nullptr, z1_bf, zall_bf, l);
  }

  // ---- fused logits / log-softmax / loss / acc (WMMA + online LSE) ----
  logits_loss_kernel<<<NROWS_ / 16, 256, 0, stream>>>(zall_bf, wout_bf, b_out, ys, accum);
  finalize_kernel<<<1, 32, 0, stream>>>(accum, out);
}
